// SelectiveScan_86414741996283
// MI455X (gfx1250) — matrix operand order repaired and checked by reading, not compiled, not run on hardware
//
#include <hip/hip_runtime.h>
#include <hip/hip_bf16.h>
#include <stdint.h>

// ---------------------------------------------------------------------------
// Mamba selective-scan forward on MI455X (gfx1250, wave32):
//   GEMM1 (bf16 WMMA) -> fused conv/silu/softplus/scan/gate -> GEMM2 (bf16 WMMA)
// GEMM staging uses CDNA5 async global->LDS DMA (ASYNCcnt) instead of
// VGPR round-trips, double-buffered against the WMMA pipeline.
// ---------------------------------------------------------------------------

typedef __attribute__((ext_vector_type(16))) __bf16 v16bf;
typedef __attribute__((ext_vector_type(8)))  __bf16 v8bf;
typedef __attribute__((ext_vector_type(8)))  float  v8f;

#define DIMM    768
#define DSTATE  16
#define DINNER  1536
#define LSEQ    4096
#define NXZ     4640   // 3*DINNER + 2*DSTATE

// GEMM tiling: block tile 128x128x32, 8 wave32s (4 along M x 2 along N),
// each wave computes a 32x64 macro-tile = 2x4 grid of 16x16x32 WMMAs.
#define BM 128
#define BN 128
#define BK 32
#define LDSS 40   // padded per-row k-stride in bf16 elements (80B, bank-friendly)

__device__ __forceinline__ __bf16 f32_to_bf16(float f) {
  union { float f; unsigned u; } v; v.f = f;
  unsigned r = v.u + 0x7FFFu + ((v.u >> 16) & 1u);   // round-to-nearest-even
  unsigned short h = (unsigned short)(r >> 16);
  __bf16 out;
  __builtin_memcpy(&out, &h, 2);
  return out;
}

__global__ __launch_bounds__(256)
void cvt_f32_to_bf16_kernel(const float* __restrict__ src,
                            __bf16* __restrict__ dst, int n) {
  int i = blockIdx.x * 256 + threadIdx.x;
  if (i < n) dst[i] = f32_to_bf16(src[i]);
}

// Issue one 32B (2 x b128) async global->LDS copy. The instruction's offset
// immediate is added to BOTH the LDS and global addresses (ISA 10.7/15.18),
// so +16 covers the second 16B chunk on both sides.
__device__ __forceinline__ void async_copy_32B(unsigned lds_addr, const void* gptr) {
  asm volatile(
      "global_load_async_to_lds_b128 %0, %1, off\n\t"
      "global_load_async_to_lds_b128 %0, %1, off offset:16"
      :
      : "v"(lds_addr), "v"(gptr)
      : "memory");
}

__device__ __forceinline__ void async_wait0() {
  asm volatile("s_wait_asynccnt 0x0" ::: "memory");
}

// C[M,N] = A[M,K] @ B[N,K]^T, bf16 inputs, fp32 accumulate/output.
__global__ __launch_bounds__(256)
void wmma_gemm_bf16_nt(const __bf16* __restrict__ A,
                       const __bf16* __restrict__ Bm,
                       float* __restrict__ C,
                       int M, int N, int K) {
  __shared__ __align__(16) __bf16 sA[2][BM][LDSS];
  __shared__ __align__(16) __bf16 sB[2][BN][LDSS];

  const int tid  = threadIdx.x;
  const int lane = tid & 31;
  const int wave = tid >> 5;
  const int wm   = wave & 3;    // 4 waves along M
  const int wn   = wave >> 2;   // 2 waves along N
  const int blockM = blockIdx.y * BM;
  const int blockN = blockIdx.x * BN;

  v8f acc[2][4];
  {
    v8f vzero = {};
#pragma unroll
    for (int i = 0; i < 2; ++i)
#pragma unroll
      for (int j = 0; j < 4; ++j) acc[i][j] = vzero;
  }

  // Cooperative tile loader: 256 threads, each DMAs 16 bf16 (32B) per matrix.
  const int lrow  = tid & 127;
  const int lhalf = tid >> 7;                       // k half: 0 -> k[0..15], 1 -> k[16..31]
  const int gcB   = (blockN + lrow < N) ? (blockN + lrow) : (N - 1);  // clamp ragged N

  auto load_tile_async = [&](int ks, int buf) {
    const int gk = ks * BK + lhalf * 16;
    const __bf16* gA = A + (size_t)(blockM + lrow) * K + gk;
    const __bf16* gB = Bm + (size_t)gcB * K + gk;
    const unsigned ldsA = (unsigned)(uintptr_t)&sA[buf][lrow][lhalf * 16];
    const unsigned ldsB = (unsigned)(uintptr_t)&sB[buf][lrow][lhalf * 16];
    async_copy_32B(ldsA, gA);
    async_copy_32B(ldsB, gB);
  };

  load_tile_async(0, 0);
  async_wait0();
  __syncthreads();

  const int nk    = K / BK;
  const int rA    = lane & 15;
  const int kloA  = (lane < 16) ? 0 : 16;   // contiguous k: lane half h holds k = 16h..16h+15
  const int kloB  = (lane < 16) ? 0 : 16;   // contiguous k, same order as A: lane half h holds k = 16h..16h+15

  for (int ks = 0; ks < nk; ++ks) {
    const int buf = ks & 1;
    if (ks + 1 < nk) load_tile_async(ks + 1, buf ^ 1);
    if (ks + 2 < nk) {   // global_prefetch_b8 for tile after next
      const int gk = (ks + 2) * BK + lhalf * 16;
      __builtin_prefetch(A + (size_t)(blockM + lrow) * K + gk, 0, 1);
      __builtin_prefetch(Bm + (size_t)gcB * K + gk, 0, 1);
    }

    v16bf afrag[2];
#pragma unroll
    for (int i = 0; i < 2; ++i) {
      const __bf16* p = &sA[buf][wm * 32 + i * 16 + rA][kloA];
      v8bf lo = *(const v8bf*)(p);        // K = klo..klo+7
      v8bf hi = *(const v8bf*)(p + 8);    // K = klo+8..klo+15
      afrag[i] = __builtin_shufflevector(lo, hi,
          0, 1, 2, 3, 4, 5, 6, 7, 8, 9, 10, 11, 12, 13, 14, 15);
    }
    v16bf bfrag[4];
#pragma unroll
    for (int j = 0; j < 4; ++j) {
      const __bf16* p = &sB[buf][wn * 64 + j * 16 + rA][kloB];
      v8bf lo = *(const v8bf*)(p);        // K = klo..klo+7
      v8bf hi = *(const v8bf*)(p + 8);    // K = klo+8..klo+15
      bfrag[j] = __builtin_shufflevector(lo, hi,
          0, 1, 2, 3, 4, 5, 6, 7, 8, 9, 10, 11, 12, 13, 14, 15);
    }

#pragma unroll
    for (int i = 0; i < 2; ++i)
#pragma unroll
      for (int j = 0; j < 4; ++j)
        acc[i][j] = __builtin_amdgcn_wmma_f32_16x16x32_bf16(
            false, afrag[i], false, bfrag[j], (short)0, acc[i][j], false, false);

    // Next tile's DMA was issued before the 8 WMMAs; by now it should be
    // complete or nearly so. Drain ASYNCcnt, then barrier before buffer flip.
    async_wait0();
    __syncthreads();
  }

  // C/D 16x16 fp32 layout: VGPR v holds M=v (lanes 0-15) / M=v+8 (lanes 16-31).
  const int rowOff  = (lane < 16) ? 0 : 8;
  const int colLane = lane & 15;
#pragma unroll
  for (int i = 0; i < 2; ++i) {
#pragma unroll
    for (int j = 0; j < 4; ++j) {
      const int cN = blockN + wn * 64 + j * 16 + colLane;
      if (cN < N) {
        const int rBase = blockM + wm * 32 + i * 16 + rowOff;
#pragma unroll
        for (int v = 0; v < 8; ++v)
          C[(size_t)(rBase + v) * N + cN] = acc[i][j][v];
      }
    }
  }
}

// Fused depthwise conv(k=3)+SiLU, softplus(delta), selective scan, D-skip, z-gate.
// One lane per (channel, state): wave32 = 2 channels x 16 states.
// Block = 256 threads = 16 channels; grid = 1536/16 = 96 blocks.
__global__ __launch_bounds__(256)
void mamba_scan_kernel(const float* __restrict__ xz,      // [LSEQ, NXZ]
                       const float* __restrict__ Wconv,   // [DINNER, 3]
                       const float* __restrict__ A_log,   // [DINNER, 16]
                       const float* __restrict__ Dskip,   // [DINNER]
                       __bf16* __restrict__ y_out) {      // [LSEQ, DINNER] bf16
  const int s = threadIdx.x & 15;                 // state index
  const int c = blockIdx.x * 16 + (threadIdx.x >> 4);  // channel index

  const float Acs = -__expf(A_log[c * DSTATE + s]);
  const float w0 = Wconv[c * 3 + 0];
  const float w1 = Wconv[c * 3 + 1];
  const float w2 = Wconv[c * 3 + 2];
  const float Dv = Dskip[c];

  float h = 0.f;
  float xprev = 0.f;
  float xcur = xz[(size_t)0 * NXZ + c];

  for (int t = 0; t < LSEQ; ++t) {
    const float* row = xz + (size_t)t * NXZ;
    const float xnext = (t + 1 < LSEQ) ? xz[(size_t)(t + 1) * NXZ + c] : 0.f;
    const float d_raw = row[2 * DINNER + 2 * DSTATE + c];   // delta
    const float Bt    = row[2 * DINNER + s];                // B_ssm
    const float Ct    = row[2 * DINNER + DSTATE + s];       // C_ssm
    const float zt    = row[DINNER + c];                    // z

    // softplus(delta)
    const float dt = (d_raw > 20.f) ? d_raw : __logf(1.f + __expf(d_raw));
    // depthwise conv + SiLU
    const float xconv = w0 * xprev + w1 * xcur + w2 * xnext;
    const float xc = xconv / (1.f + __expf(-xconv));
    // state update
    const float dA = __expf(dt * Acs);
    h = dA * h + dt * Bt * xc;
    float yp = h * Ct;
    // butterfly reduce over 16 states (stays within the channel's lane group)
    yp += __shfl_xor(yp, 1, 32);
    yp += __shfl_xor(yp, 2, 32);
    yp += __shfl_xor(yp, 4, 32);
    yp += __shfl_xor(yp, 8, 32);

    if (s == 0) {
      const float y = yp + xc * Dv;
      const float g = zt / (1.f + __expf(-zt));   // silu(z)
      y_out[(size_t)t * DINNER + c] = f32_to_bf16(y * g);
    }
    xprev = xcur;
    xcur = xnext;
  }
}

extern "C" void kernel_launch(void* const* d_in, const int* in_sizes, int n_in,
                              void* d_out, int out_size, void* d_ws, size_t ws_size,
                              hipStream_t stream) {
  (void)in_sizes; (void)n_in; (void)out_size; (void)ws_size;
  const float* x      = (const float*)d_in[0];   // [1, 4096, 768]
  const float* W_in   = (const float*)d_in[1];   // [4640, 768]
  const float* W_conv = (const float*)d_in[2];   // [1536, 1, 3]
  const float* W_out  = (const float*)d_in[3];   // [768, 1536]
  const float* A_log  = (const float*)d_in[4];   // [1536, 16]
  const float* Dskip  = (const float*)d_in[5];   // [1536]
  float* out = (float*)d_out;                    // [4096, 768] fp32

  // Workspace layout (256B aligned slabs).
  char* ws = (char*)d_ws;
  size_t off = 0;
  auto take = [&](size_t bytes) -> char* {
    char* p = ws + off;
    off = (off + bytes + 255) & ~(size_t)255;
    return p;
  };
  __bf16* x_bf    = (__bf16*)take((size_t)LSEQ * DIMM * 2);
  __bf16* win_bf  = (__bf16*)take((size_t)NXZ * DIMM * 2);
  __bf16* wout_bf = (__bf16*)take((size_t)DIMM * DINNER * 2);
  float*  xz      = (float*) take((size_t)LSEQ * NXZ * 4);
  __bf16* y_bf    = (__bf16*)take((size_t)LSEQ * DINNER * 2);

  // 1) fp32 -> bf16 conversions
  {
    int n = LSEQ * DIMM;
    cvt_f32_to_bf16_kernel<<<(n + 255) / 256, 256, 0, stream>>>(x, x_bf, n);
  }
  {
    int n = NXZ * DIMM;
    cvt_f32_to_bf16_kernel<<<(n + 255) / 256, 256, 0, stream>>>(W_in, win_bf, n);
  }
  {
    int n = DIMM * DINNER;
    cvt_f32_to_bf16_kernel<<<(n + 255) / 256, 256, 0, stream>>>(W_out, wout_bf, n);
  }

  // 2) GEMM1: xz[4096,4640] = x_bf @ win_bf^T  (K=768)
  {
    dim3 grid((NXZ + BN - 1) / BN, LSEQ / BM);
    wmma_gemm_bf16_nt<<<grid, 256, 0, stream>>>(x_bf, win_bf, xz, LSEQ, NXZ, DIMM);
  }

  // 3) fused conv/silu/softplus/scan/gate -> y_bf[4096,1536]
  mamba_scan_kernel<<<DINNER / 16, 256, 0, stream>>>(xz, W_conv, A_log, Dskip, y_bf);

  // 4) GEMM2: out[4096,768] = y_bf @ wout_bf^T  (K=1536)
  {
    dim3 grid(DIMM / BN, LSEQ / BM);
    wmma_gemm_bf16_nt<<<grid, 256, 0, stream>>>(y_bf, wout_bf, out, LSEQ, DIMM, DINNER);
  }
}
